// Fast_RCNN_89404039234173
// MI455X (gfx1250) — matrix operand order repaired and checked by reading, not compiled, not run on hardware
//
#include <hip/hip_runtime.h>
#include <hip/hip_bf16.h>

typedef __bf16 bf16_t;
typedef __attribute__((ext_vector_type(16))) __bf16 v16bf;
typedef __attribute__((ext_vector_type(8)))  __bf16 v8bf;
typedef __attribute__((ext_vector_type(8)))  float  v8f;
typedef __attribute__((ext_vector_type(2)))  float  v2f;
typedef __attribute__((ext_vector_type(4)))  unsigned int v4u;
typedef __attribute__((ext_vector_type(8)))  int    v8i;
typedef __attribute__((ext_vector_type(4)))  int    v4i;

#define ACT_STRIDE 3211264ull   // 64*224*224 elems: max per-image activation footprint

// ---------------------------------------------------------------------------
// Tensor Data Mover: 2D tile load Global->LDS (gfx1250 only).
// D# per CDNA5 ISA ch.8: group0 {count=1, lds_addr, global_addr[56:0], type=2},
// group1 {data_size, tensor_dim0/1 (relative to tile start -> OOB zero-fill),
//         tile_dim0/1, tensor_dim0_stride}. Groups 2/3 zero (<=2D tensor).
// ---------------------------------------------------------------------------
#if defined(__gfx1250__)
__device__ __forceinline__ void tdm_load_2d(unsigned lds_off, const void* gaddr,
                                            unsigned dsize_code,      // 1=2B, 2=4B
                                            unsigned tensor_d0, unsigned tensor_d1,
                                            unsigned tile_d0, unsigned tile_d1,
                                            unsigned long long stride0_elems) {
    unsigned long long ga = (unsigned long long)(uintptr_t)gaddr;
    v4u g0;
    g0.x = 1u;                                               // count=1, user D#
    g0.y = lds_off;                                          // lds_addr (bytes)
    g0.z = (unsigned)(ga & 0xffffffffu);                     // global_addr[31:0]
    g0.w = (unsigned)((ga >> 32) & 0x01ffffffu) | (2u << 30);// addr[56:32] | type=2
    v8i g1;
    g1[0] = (int)(dsize_code << 16);                         // wg_mask=0 | data_size
    g1[1] = (int)((tensor_d0 & 0xffffu) << 16);              // abar=0 | dim0[15:0]
    g1[2] = (int)((tensor_d0 >> 16) | ((tensor_d1 & 0xffffu) << 16));
    g1[3] = (int)((tensor_d1 >> 16) | (tile_d0 << 16));      // dim1[31:16] | tile0
    g1[4] = (int)(tile_d1 & 0xffffu);                        // tile1 | tile2=0
    g1[5] = (int)(stride0_elems & 0xffffffffu);
    g1[6] = (int)((stride0_elems >> 32) & 0xffffu);          // stride0 hi | stride1=0
    g1[7] = 0;
    v4i gz = {0, 0, 0, 0};
#if __clang_major__ >= 23
    v8i gz8 = {0, 0, 0, 0, 0, 0, 0, 0};
    __builtin_amdgcn_tensor_load_to_lds(g0, g1, gz, gz, gz8, 0);
#else
    __builtin_amdgcn_tensor_load_to_lds(g0, g1, gz, gz, 0);
#endif
}
#endif

// ---------------------------------------------------------------------------
// Fold BN into conv weights: w' = w*scale[cout] (bf16), b' = b*scale+shift
// ---------------------------------------------------------------------------
__global__ void fold_weights(const float* __restrict__ w, const float* __restrict__ b,
                             const float* __restrict__ sc, const float* __restrict__ sh,
                             bf16_t* __restrict__ wout, float* __restrict__ bout,
                             int cout, int k) {
    int i = blockIdx.x * blockDim.x + threadIdx.x;
    int n = cout * k;
    if (i < n) {
        int co = i / k;
        wout[i] = (bf16_t)(w[i] * sc[co]);
    }
    if (i < cout) bout[i] = b[i] * sc[i] + sh[i];
}

__global__ void img_to_bf16(const float* __restrict__ in, bf16_t* __restrict__ out, int perImg) {
    int img = blockIdx.y;
    int i = blockIdx.x * blockDim.x + threadIdx.x;
    if (i < perImg) out[img * ACT_STRIDE + i] = (bf16_t)in[(size_t)img * perImg + i];
}

// ---------------------------------------------------------------------------
// 3x3 SAME conv as implicit GEMM, bf16 WMMA 16x16x32, f32 accum, fused
// bias+ReLU. Weight tile via TDM (OOB zero-fill pads K). im2col staging:
// contiguous pixels per wave (coalesced) with readfirstlane-scalarized k
// decomposition (div-by-9/3 and tile offset live in SALU).
// Block = 256 thr (8 waves). Tile: M=64 couts x N=64 pixels, K step 32.
// ---------------------------------------------------------------------------
__global__ __launch_bounds__(256)
void conv3x3_wmma(const bf16_t* __restrict__ act_in, const bf16_t* __restrict__ wgt,
                  const float* __restrict__ bias, bf16_t* __restrict__ act_out,
                  int Cin, int Cout, int H, int W) {
    const int HW = H * W;
    const int K  = Cin * 9;
    const int img = blockIdx.z;
    const bf16_t* in  = act_in  + (size_t)img * ACT_STRIDE;
    bf16_t*       out = act_out + (size_t)img * ACT_STRIDE;
    const int n0 = blockIdx.x * 64;   // pixel tile base
    const int m0 = blockIdx.y * 64;   // cout tile base

    __shared__ bf16_t Alds[64 * 32];  // [m][k]   <- TDM target
    __shared__ bf16_t Blds[64 * 32];  // [pixel][k] (column-major B)

    const int tid  = threadIdx.x;
    const int wave = tid >> 5;
    const int lane = tid & 31;
    const int wm   = wave >> 1;
    const int wn   = wave & 1;
    const int hl   = lane >> 4;
    const int ml   = lane & 15;

    // im2col staging geometry: fixed pixel per thread, wave-uniform k group
    const int p    = tid & 63;        // pixel within tile
    const int kgrp = __builtin_amdgcn_readfirstlane(tid >> 6); // 0..3, SALU-uniform
    const int pix  = n0 + p;
    const bool pok = pix < HW;
    const int py   = pok ? (pix / W) : 0;
    const int px   = pix - py * W;
    const int tb   = (py - 1) * W + (px - 1);  // per-thread base offset

#if defined(__gfx1250__)
    const unsigned alds_off = (unsigned)(uintptr_t)(void*)&Alds[0];
#endif

    v8f acc0 = {}; v8f acc1 = {};

    for (int k0 = 0; k0 < K; k0 += 32) {
        __syncthreads();
#if defined(__gfx1250__)
        if (wave == 0)   // 64x32 bf16 tile, row stride K; OOB (k>=K) zero-fills
            tdm_load_2d(alds_off, wgt + (size_t)m0 * K + k0, /*2B*/1,
                        (unsigned)(K - k0), (unsigned)(Cout - m0),
                        32u, 64u, (unsigned long long)K);
#else
        for (int i = tid; i < 64 * 8; i += 256) {
            int r = i >> 3, kq = (i & 7) << 2;
            for (int j = 0; j < 4; ++j) {
                int gk = k0 + kq + j;
                Alds[r * 32 + kq + j] = (gk < K) ? wgt[(size_t)(m0 + r) * K + gk]
                                                 : (bf16_t)0.f;
            }
        }
#endif
        // ---- im2col B tile: wave-uniform (scalar) k, contiguous pixels ----
        #pragma unroll
        for (int t = 0; t < 8; ++t) {
            const int kk = kgrp + 4 * t;                 // scalar
            const int gk = k0 + kk;                      // scalar
            bf16_t v = (bf16_t)0.f;
            if (gk < K) {                                // scalar branch
                const int ci = gk / 9, rs = gk - ci * 9; // SALU
                const int rr = rs / 3, ss = rs - rr * 3; // SALU
                const int soff = ci * HW + rr * W + ss;  // SALU
                const int iy = py + rr - 1, ix = px + ss - 1;
                if (pok && iy >= 0 && iy < H && ix >= 0 && ix < W)
                    v = in[soff + tb];                   // coalesced over p
            }
            Blds[p * 32 + kk] = v;
        }
#if defined(__gfx1250__)
        if (wave == 0) __builtin_amdgcn_s_wait_tensorcnt(0);
#endif
        __syncthreads();

        // ---- A fragment: lanes<16 K{0..15}, lanes>=16 K{16..31} ----
        v16bf af;
        {
            const bf16_t* ap = &Alds[(wm * 16 + ml) * 32 + hl * 16];
            v8bf lo = *(const v8bf*)(ap);
            v8bf hi = *(const v8bf*)(ap + 8);
            #pragma unroll
            for (int j = 0; j < 8; ++j) { af[j] = lo[j]; af[8 + j] = hi[j]; }
        }
        v16bf bf0, bf1;
        {
            const bf16_t* bp = &Blds[(wn * 32 + ml) * 32 + hl * 16];
            v8bf lo = *(const v8bf*)(bp);
            v8bf hi = *(const v8bf*)(bp + 8);
            #pragma unroll
            for (int j = 0; j < 8; ++j) { bf0[j] = lo[j]; bf0[8 + j] = hi[j]; }
        }
        {
            const bf16_t* bp = &Blds[(wn * 32 + 16 + ml) * 32 + hl * 16];
            v8bf lo = *(const v8bf*)(bp);
            v8bf hi = *(const v8bf*)(bp + 8);
            #pragma unroll
            for (int j = 0; j < 8; ++j) { bf1[j] = lo[j]; bf1[8 + j] = hi[j]; }
        }
        acc0 = __builtin_amdgcn_wmma_f32_16x16x32_bf16(false, af, false, bf0,
                                                       (short)0, acc0, false, false);
        acc1 = __builtin_amdgcn_wmma_f32_16x16x32_bf16(false, af, false, bf1,
                                                       (short)0, acc1, false, false);
    }

    #pragma unroll
    for (int v = 0; v < 8; ++v) {
        int m  = m0 + wm * 16 + 8 * hl + v;
        int p0 = n0 + wn * 32 + ml;
        float bz = bias[m];
        float r0 = fmaxf(acc0[v] + bz, 0.f);
        float r1 = fmaxf(acc1[v] + bz, 0.f);
        if (p0 < HW)      out[m * HW + p0]      = (bf16_t)r0;
        if (p0 + 16 < HW) out[m * HW + p0 + 16] = (bf16_t)r1;
    }
}

__global__ void maxpool2(const bf16_t* __restrict__ in, bf16_t* __restrict__ out,
                         int C, int H, int W) {
    int img = blockIdx.y;
    const bf16_t* ip = in  + (size_t)img * ACT_STRIDE;
    bf16_t*       op = out + (size_t)img * ACT_STRIDE;
    int oh = H >> 1, ow = W >> 1;
    int total = C * oh * ow;
    int i = blockIdx.x * blockDim.x + threadIdx.x;
    if (i >= total) return;
    int c = i / (oh * ow);
    int rem = i - c * oh * ow;
    int y = rem / ow, x = rem - y * ow;
    const bf16_t* pp = ip + c * H * W + (2 * y) * W + 2 * x;
    float m = fmaxf(fmaxf((float)pp[0], (float)pp[1]),
                    fmaxf((float)pp[W], (float)pp[W + 1]));
    op[i] = (bf16_t)m;
}

__global__ void roi_pool(const bf16_t* __restrict__ feat, const int* __restrict__ regions,
                         float* __restrict__ pooled) {
    int r = blockIdx.x;            // 0..127
    int img = r >> 6;
    const int* reg = regions + r * 4;
    const float s = 14.f / 224.f;
    int c0 = (int)rintf(reg[0] * s);
    int c1 = (int)rintf(reg[1] * s);
    int c2 = (int)rintf(reg[2] * s);
    int c3 = (int)rintf(reg[3] * s);
    int x0 = min(max(c0, 0), 13);
    int y0 = min(max(c1, 0), 13);
    int x1 = max(min(c2, 13), x0);
    int y1 = max(min(c3, 13), y0);
    int Hh = x1 - x0 + 1, Ww = y1 - y0 + 1;

    int t = blockIdx.y * blockDim.x + threadIdx.x;
    if (t >= 512 * 49) return;
    int c = t / 49;
    int ij = t - c * 49;
    int i = ij / 7, j = ij - i * 7;
    int sh_ = x0 + (i * Hh) / 7;
    int eh_ = x0 + ((i + 1) * Hh + 6) / 7;
    int sw_ = y0 + (j * Ww) / 7;
    int ew_ = y0 + ((j + 1) * Ww + 6) / 7;
    const bf16_t* fp = feat + img * ACT_STRIDE + c * 196;
    float m = -1e30f;
    for (int h = sh_; h < eh_; ++h)
        for (int w = sw_; w < ew_; ++w)
            m = fmaxf(m, (float)fp[h * 14 + w]);
    pooled[(size_t)r * 25088 + c * 49 + ij] = m;
}

// ---------------------------------------------------------------------------
// FC GEMM: Out[128][N] = relu(A[128][K] x W[K][N] + bias). Memory-bound on W
// (fc_w0 = 411MB @ 23.3TB/s) -> stream f32 weights exactly once via TDM,
// double-buffered in LDS (TENSORcnt-pipelined), f32 WMMA 16x16x4.
// Block = 8 waves, tile M=128 x N=64. K multiple of 32 (25088, 4096).
// ---------------------------------------------------------------------------
__global__ __launch_bounds__(256)
void fc_wmma(const float* __restrict__ A, const float* __restrict__ Wt,
             const float* __restrict__ bias, float* __restrict__ Out,
             int K, int N) {
    const int nb0 = blockIdx.x * 64;
    __shared__ float Al[128 * 32];      // [m][k]
    __shared__ float Bl[2][32 * 64];    // [k][n], double-buffered TDM target
    const int tid  = threadIdx.x;
    const int wave = tid >> 5;
    const int lane = tid & 31;
    const int m0   = wave * 16;
    const int hl   = lane >> 4;
    const int ml   = lane & 15;

    v8f acc[4] = {{}, {}, {}, {}};
    const int nsteps = K / 32;

#if defined(__gfx1250__)
    const unsigned bl_off[2] = {(unsigned)(uintptr_t)(void*)&Bl[0][0],
                                (unsigned)(uintptr_t)(void*)&Bl[1][0]};
    if (wave == 0)   // prologue: tile 0 in flight
        tdm_load_2d(bl_off[0], Wt + nb0, /*4B*/2,
                    (unsigned)(N - nb0), (unsigned)K, 64u, 32u,
                    (unsigned long long)N);
#endif

    for (int s = 0; s < nsteps; ++s) {
        const int k0  = s * 32;
        const int cur = s & 1;
        __syncthreads();                 // prev compute done: Al & Bl[!cur] free
        // stage A tile 128x32 (small, regular loads)
        for (int i = tid; i < 1024; i += 256) {
            int r = i >> 3, kq = (i & 7) << 2;
            *(float4*)&Al[r * 32 + kq] = *(const float4*)&A[(size_t)r * K + k0 + kq];
        }
#if defined(__gfx1250__)
        if (wave == 0) {
            if (s + 1 < nsteps) {        // issue next tile, then wait for current
                tdm_load_2d(bl_off[cur ^ 1], Wt + (size_t)(k0 + 32) * N + nb0, 2,
                            (unsigned)(N - nb0), (unsigned)(K - (k0 + 32)),
                            64u, 32u, (unsigned long long)N);
                __builtin_amdgcn_s_wait_tensorcnt(1);
            } else {
                __builtin_amdgcn_s_wait_tensorcnt(0);
            }
        }
        if (k0 + 32 < K)                 // gfx1250 global_prefetch_b8: next A slab
            __builtin_prefetch((const void*)&A[k0 + 32], 0, 0);
#else
        {
            float* Blc = &Bl[cur][0];
            for (int i = tid; i < 512; i += 256) {
                int r = i >> 4, nq = (i & 15) << 2;
                *(float4*)&Blc[r * 64 + nq] =
                    *(const float4*)&Wt[(size_t)(k0 + r) * N + nb0 + nq];
            }
        }
#endif
        __syncthreads();                 // Al staged + Bl[cur] landed

        const float* Blc = &Bl[cur][0];
        #pragma unroll
        for (int k4 = 0; k4 < 32; k4 += 4) {
            v2f a;
            a[0] = Al[(m0 + ml) * 32 + k4 + 2 * hl];
            a[1] = Al[(m0 + ml) * 32 + k4 + 2 * hl + 1];
            #pragma unroll
            for (int nt = 0; nt < 4; ++nt) {
                v2f b;
                b[0] = Blc[(k4 + 2 * hl) * 64 + nt * 16 + ml];
                b[1] = Blc[(k4 + 2 * hl + 1) * 64 + nt * 16 + ml];
                acc[nt] = __builtin_amdgcn_wmma_f32_16x16x4_f32(false, a, false, b,
                                                                (short)0, acc[nt],
                                                                false, false);
            }
        }
    }

    #pragma unroll
    for (int nt = 0; nt < 4; ++nt) {
        #pragma unroll
        for (int v = 0; v < 8; ++v) {
            int m = m0 + 8 * hl + v;
            int n = nb0 + nt * 16 + ml;
            Out[(size_t)m * N + n] = fmaxf(acc[nt][v] + bias[n], 0.f);
        }
    }
}

// ---------------------------------------------------------------------------
// Host-side orchestration
// ---------------------------------------------------------------------------
static const int  CIN_[13]  = {3, 64, 64, 128, 128, 256, 256, 256, 512, 512, 512, 512, 512};
static const int  COUT_[13] = {64, 64, 128, 128, 256, 256, 256, 512, 512, 512, 512, 512, 512};
static const bool PAFT_[13] = {false, true, false, true, false, false, true,
                               false, false, true, false, false, false};

extern "C" void kernel_launch(void* const* d_in, const int* in_sizes, int n_in,
                              void* d_out, int out_size, void* d_ws, size_t ws_size,
                              hipStream_t stream) {
    const float* inputs  = (const float*)d_in[0];
    const int*   regions = (const int*)d_in[1];
    const float* convw[13]; const float* convb[13];
    const float* bnsc[13]; const float* bnsh[13];
    for (int i = 0; i < 13; ++i) {
        convw[i] = (const float*)d_in[2 + i];
        convb[i] = (const float*)d_in[15 + i];
        bnsc[i]  = (const float*)d_in[28 + i];
        bnsh[i]  = (const float*)d_in[41 + i];
    }
    const float* fcw0 = (const float*)d_in[54];
    const float* fcw1 = (const float*)d_in[55];
    const float* fcb0 = (const float*)d_in[56];
    const float* fcb1 = (const float*)d_in[57];

    char* ws = (char*)d_ws;
    size_t off = 0;
    auto alignup = [](size_t x) { return (x + 255) & ~(size_t)255; };
    bf16_t* wbf   = (bf16_t*)(ws + off); off = alignup(off + 14710464ull * sizeof(bf16_t));
    float*  biasf = (float*)(ws + off);  off = alignup(off + 4224ull * sizeof(float));
    bf16_t* actA  = (bf16_t*)(ws + off); off = alignup(off + 2 * ACT_STRIDE * sizeof(bf16_t));
    bf16_t* actB  = (bf16_t*)(ws + off); off = alignup(off + 2 * ACT_STRIDE * sizeof(bf16_t));
    float*  pooled= (float*)(ws + off);  off = alignup(off + 128ull * 25088 * sizeof(float));
    float*  h1    = (float*)(ws + off);  off = alignup(off + 128ull * 4096 * sizeof(float));

    {   // 1) fold BN
        size_t woff = 0, boff = 0;
        for (int i = 0; i < 13; ++i) {
            int k = CIN_[i] * 9, n = COUT_[i] * k;
            fold_weights<<<(n + 255) / 256, 256, 0, stream>>>(
                convw[i], convb[i], bnsc[i], bnsh[i],
                wbf + woff, biasf + boff, COUT_[i], k);
            woff += (size_t)n; boff += COUT_[i];
        }
    }
    {   // 2) input -> bf16
        int perImg = 3 * 224 * 224;
        dim3 g((perImg + 255) / 256, 2);
        img_to_bf16<<<g, 256, 0, stream>>>(inputs, actA, perImg);
    }
    // 3) backbone
    bf16_t* cur = actA; bf16_t* nxt = actB;
    int sp = 224;
    size_t woff = 0, boff = 0;
    for (int i = 0; i < 13; ++i) {
        int HW = sp * sp;
        dim3 g((HW + 63) / 64, COUT_[i] / 64, 2);
        conv3x3_wmma<<<g, 256, 0, stream>>>(cur, wbf + woff, biasf + boff, nxt,
                                            CIN_[i], COUT_[i], sp, sp);
        { bf16_t* t = cur; cur = nxt; nxt = t; }
        if (PAFT_[i]) {
            int oh = sp / 2;
            int total = COUT_[i] * oh * oh;
            dim3 gp((total + 255) / 256, 2);
            maxpool2<<<gp, 256, 0, stream>>>(cur, nxt, COUT_[i], sp, sp);
            { bf16_t* t = cur; cur = nxt; nxt = t; }
            sp = oh;
        }
        woff += (size_t)COUT_[i] * CIN_[i] * 9; boff += COUT_[i];
    }
    // 4) RoI pool
    {
        dim3 g(128, (512 * 49 + 255) / 256);
        roi_pool<<<g, 256, 0, stream>>>(cur, regions, pooled);
    }
    // 5) FC1 + ReLU
    fc_wmma<<<4096 / 64, 256, 0, stream>>>(pooled, fcw0, fcb0, h1, 25088, 4096);
    // 6) FC2 + ReLU -> d_out
    fc_wmma<<<4096 / 64, 256, 0, stream>>>(h1, fcw1, fcb1, (float*)d_out, 4096, 4096);
}
